// MambaLiteBlock_29618094473398
// MI455X (gfx1250) — matrix operand order repaired and checked by reading, not compiled, not run on hardware
//
#include <hip/hip_runtime.h>

// ---------------- problem constants ----------------
#define B_   8
#define T_   4096
#define D_   1024
#define S_   64
#define CONVK 3
#define DS_  1088            // D + S
#define TWO_D 2048
#define M_TOT 32768          // B*T
#define NCH  64              // scan chunks
#define CHL  64              // chunk length (T / NCH)
#define EPSLN 1e-5f

typedef __bf16 bf16_t;
typedef __attribute__((ext_vector_type(16))) __bf16 v16bf;
typedef __attribute__((ext_vector_type(8)))  __bf16 v8bf;
typedef __attribute__((ext_vector_type(4)))  __bf16 v4bf;
typedef __attribute__((ext_vector_type(8)))  float  v8f;

// ---------------- CDNA5 async global->LDS path (guarded) ----------------
#if defined(__has_builtin)
#if __has_builtin(__builtin_amdgcn_global_load_async_to_lds_b128)
#define USE_ASYNC 1
#endif
#endif
#ifndef USE_ASYNC
#define USE_ASYNC 0
#endif

#if USE_ASYNC
typedef int a4i __attribute__((vector_size(16)));
typedef __attribute__((address_space(3))) a4i* lds4i_t;
#if __has_builtin(__builtin_amdgcn_s_wait_asynccnt)
#define ASYNC_WAIT() __builtin_amdgcn_s_wait_asynccnt(0)
#else
#define ASYNC_WAIT() asm volatile("s_wait_asynccnt 0x0" ::: "memory")
#endif
#else
#define ASYNC_WAIT() ((void)0)
#endif

// ---------------- WMMA fragment helpers (ISA 7.12.2 layouts, wave32) ----------------
// A 16x32 bf16 fragment from row-major global memory.
__device__ __forceinline__ v16bf load_a_frag(const bf16_t* __restrict__ A, int lda,
                                             int m0, int k0, int lane) {
    int m  = m0 + (lane & 15);
    int kb = k0 + ((lane >> 4) << 3);
    const bf16_t* p = A + (size_t)m * lda + kb;
    v8bf lo = *(const v8bf*)(p);
    v8bf hi = *(const v8bf*)(p + 16);
    v16bf r;
#pragma unroll
    for (int i = 0; i < 8; ++i) { r[i] = lo[i]; r[i + 8] = hi[i]; }
    return r;
}

// B 32x16 fragment from an LDS tile laid out [rows][64] (row = n, 64 = K-chunk)
__device__ __forceinline__ v16bf lds_b_frag(const bf16_t* buf, int n_local, int ks, int lane) {
    int n  = n_local + (lane & 15);
    int kb = ks + ((lane >> 4) << 3);
    v8bf lo = *(const v8bf*)(buf + n * 64 + kb);
    v8bf hi = *(const v8bf*)(buf + n * 64 + kb + 16);
    v16bf r;
#pragma unroll
    for (int i = 0; i < 8; ++i) { r[i] = lo[i]; r[i + 8] = hi[i]; }
    return r;
}

__device__ __forceinline__ v8f wmma_bf16(v16bf a, v16bf b, v8f c) {
    return __builtin_amdgcn_wmma_f32_16x16x32_bf16(false, a, false, b, (short)0, c,
                                                   false, false);
}

// Stage a ROWS x 64 bf16 weight tile (row-major, ldb) into LDS buf[ROWS][64].
// 16B granules; async copy (ASYNCcnt) when available, else load+ds_store.
template <int ROWS>
__device__ __forceinline__ void stage_tile(const bf16_t* __restrict__ W, int ldb, int n0,
                                           int kk, bf16_t* buf, int tid) {
    for (int i = tid; i < ROWS * 8; i += 256) {
        int row = i >> 3;
        int kc  = (i & 7) << 3;
        const bf16_t* g = W + (size_t)(n0 + row) * ldb + kk + kc;
        bf16_t* l = buf + row * 64 + kc;
#if USE_ASYNC
        __builtin_amdgcn_global_load_async_to_lds_b128((a4i*)g, (lds4i_t)l, 0, 0);
#else
        *(v8bf*)l = *(const v8bf*)g;
#endif
    }
}

// ---------------- fp32 -> bf16 repack ----------------
__global__ void f2bf_kernel(const float* __restrict__ src, bf16_t* __restrict__ dst, int n) {
    int i = blockIdx.x * blockDim.x + threadIdx.x;
    if (i < n) dst[i] = (bf16_t)src[i];
}

// ---------------- LayerNorm -> bf16 (one float4 per thread, D=1024/256) --------------
__global__ void ln_kernel(const float* __restrict__ x, const float* __restrict__ w,
                          const float* __restrict__ bsv, bf16_t* __restrict__ xn) {
    const int row = blockIdx.x;
    const int tid = threadIdx.x;
    const size_t base = (size_t)row * D_;
    float4 v = ((const float4*)(x + base))[tid];
    float sum = v.x + v.y + v.z + v.w;
    float sq  = v.x * v.x + v.y * v.y + v.z * v.z + v.w * v.w;
    __shared__ float s1[256], s2[256];
    s1[tid] = sum; s2[tid] = sq;
    __syncthreads();
#pragma unroll
    for (int off = 128; off > 0; off >>= 1) {
        if (tid < off) { s1[tid] += s1[tid + off]; s2[tid] += s2[tid + off]; }
        __syncthreads();
    }
    float mu  = s1[0] * (1.0f / D_);
    float var = s2[0] * (1.0f / D_) - mu * mu;
    float inv = rsqrtf(var + EPSLN);
    float4 w4 = ((const float4*)w)[tid];
    float4 b4 = ((const float4*)bsv)[tid];
    v4bf o;
    o[0] = (bf16_t)((v.x - mu) * inv * w4.x + b4.x);
    o[1] = (bf16_t)((v.y - mu) * inv * w4.y + b4.y);
    o[2] = (bf16_t)((v.z - mu) * inv * w4.z + b4.z);
    o[3] = (bf16_t)((v.w - mu) * inv * w4.w + b4.w);
    ((v4bf*)(xn + base))[tid] = o;
}

// ---------------- double-buffered WMMA mainloop (wave tile 16x64) ----------------
// B fragments for the whole chunk are preloaded into registers, then 8 WMMAs issue
// back-to-back (single DS drain instead of one per WMMA).
__device__ __forceinline__ void gemm_mainloop(const bf16_t* __restrict__ A, int lda,
                                              const bf16_t* __restrict__ W, int ldb,
                                              int Ktot, int m0, int n0,
                                              int lane, int tid, bf16_t* sb, v8f* acc) {
    const int nchunks = Ktot / 64;
    stage_tile<64>(W, ldb, n0, 0, sb, tid);
    ASYNC_WAIT();
    __syncthreads();
    for (int c = 0; c < nchunks; ++c) {
        const int kk = c * 64;
        v16bf a0 = load_a_frag(A, lda, m0, kk, lane);
        v16bf a1 = load_a_frag(A, lda, m0, kk + 32, lane);
        __builtin_prefetch(A + (size_t)(m0 + (lane & 15)) * lda + kk + 128, 0, 1);
        if (c + 1 < nchunks)
            stage_tile<64>(W, ldb, n0, kk + 64, sb + ((c + 1) & 1) * (64 * 64), tid);
        const bf16_t* buf = sb + (c & 1) * (64 * 64);
        v16bf b0[4], b1[4];
#pragma unroll
        for (int j = 0; j < 4; ++j) b0[j] = lds_b_frag(buf, 16 * j, 0, lane);
#pragma unroll
        for (int j = 0; j < 4; ++j) b1[j] = lds_b_frag(buf, 16 * j, 32, lane);
#pragma unroll
        for (int j = 0; j < 4; ++j) acc[j] = wmma_bf16(a0, b0[j], acc[j]);
#pragma unroll
        for (int j = 0; j < 4; ++j) acc[j] = wmma_bf16(a1, b1[j], acc[j]);
        ASYNC_WAIT();          // next tile resident before anyone proceeds
        __syncthreads();       // and everyone done reading current tile
    }
}

// ---------------- in_proj GEMM: h = xn @ in_w^T + in_b -> xproj | z ----------------
__global__ void gemm_in_kernel(const bf16_t* __restrict__ A, const bf16_t* __restrict__ W,
                               const float* __restrict__ bias,
                               bf16_t* __restrict__ xproj, bf16_t* __restrict__ zout) {
    __shared__ __align__(32) bf16_t sb[2 * 64 * 64];
    const int tid = threadIdx.x, lane = tid & 31, wave = tid >> 5;
    const int m0 = blockIdx.x * 128 + wave * 16;
    const int n0 = blockIdx.y * 64;
    v8f acc[4] = {v8f{0}, v8f{0}, v8f{0}, v8f{0}};
    gemm_mainloop(A, D_, W, D_, D_, m0, n0, lane, tid, sb, acc);
    const int l15 = lane & 15, hi = lane >> 4;
#pragma unroll
    for (int j = 0; j < 4; ++j) {
        int n = n0 + 16 * j + l15;
#pragma unroll
        for (int v = 0; v < 8; ++v) {
            int m = m0 + v + hi * 8;
            float val = acc[j][v] + bias[n];
            if (n < D_) xproj[(size_t)m * D_ + n] = (bf16_t)val;
            else        zout [(size_t)m * D_ + (n - D_)] = (bf16_t)val;
        }
    }
}

// ---------------- depthwise conv (K=3,'same') + tanh + *d -> mix[:,0:D], 8-wide -------
__global__ void conv_kernel(const bf16_t* __restrict__ xproj, const float* __restrict__ cw,
                            const float* __restrict__ cb, const float* __restrict__ dvec,
                            bf16_t* __restrict__ mix) {
    size_t idx = (size_t)blockIdx.x * blockDim.x + threadIdx.x;
    if (idx >= (size_t)M_TOT * (D_ / 8)) return;
    int d8 = (int)(idx % (D_ / 8)) * 8;
    int bt = (int)(idx / (D_ / 8));
    int t  = bt % T_;
    const bf16_t* p = xproj + (size_t)bt * D_ + d8;
    v8bf xm1 = {}, xp1 = {};
    v8bf x0 = *(const v8bf*)p;
    if (t > 0)      xm1 = *(const v8bf*)(p - D_);
    if (t < T_ - 1) xp1 = *(const v8bf*)(p + D_);
    v8bf o;
#pragma unroll
    for (int u = 0; u < 8; ++u) {
        int dd = d8 + u;
        float acc = cb[dd] + (float)xm1[u] * cw[dd * CONVK]
                          + (float)x0[u]  * cw[dd * CONVK + 1]
                          + (float)xp1[u] * cw[dd * CONVK + 2];
        o[u] = (bf16_t)(tanhf(acc) * dvec[dd]);
    }
    *(v8bf*)(mix + (size_t)bt * DS_ + d8) = o;
}

// ---------------- gate GEMM: [dt|b|c](192) = z @ GW^T, fused nonlinearity ------------
// 12 accumulators per wave; B fragments preloaded in groups of 4 to bound VGPRs.
__global__ void gates_kernel(const bf16_t* __restrict__ Z, const bf16_t* __restrict__ GW,
                             const float* __restrict__ dt_b, const float* __restrict__ b_b,
                             const float* __restrict__ c_b, const float* __restrict__ a_log,
                             float* __restrict__ decay, float* __restrict__ bval,
                             float* __restrict__ cval) {
    __shared__ __align__(32) bf16_t sb[2 * 192 * 64];
    const int tid = threadIdx.x, lane = tid & 31, wave = tid >> 5;
    const int m0 = blockIdx.x * 128 + wave * 16;
    v8f acc[12];
#pragma unroll
    for (int j = 0; j < 12; ++j) acc[j] = v8f{0};
    const int nchunks = D_ / 64;
    stage_tile<192>(GW, D_, 0, 0, sb, tid);
    ASYNC_WAIT();
    __syncthreads();
    for (int c = 0; c < nchunks; ++c) {
        const int kk = c * 64;
        v16bf a0 = load_a_frag(Z, D_, m0, kk, lane);
        v16bf a1 = load_a_frag(Z, D_, m0, kk + 32, lane);
        if (c + 1 < nchunks)
            stage_tile<192>(GW, D_, 0, kk + 64, sb + ((c + 1) & 1) * (192 * 64), tid);
        const bf16_t* buf = sb + (c & 1) * (192 * 64);
#pragma unroll
        for (int g = 0; g < 3; ++g) {
            v16bf bb[4];
#pragma unroll
            for (int j = 0; j < 4; ++j) bb[j] = lds_b_frag(buf, 64 * g + 16 * j, 0, lane);
#pragma unroll
            for (int j = 0; j < 4; ++j) acc[4 * g + j] = wmma_bf16(a0, bb[j], acc[4 * g + j]);
#pragma unroll
            for (int j = 0; j < 4; ++j) bb[j] = lds_b_frag(buf, 64 * g + 16 * j, 32, lane);
#pragma unroll
            for (int j = 0; j < 4; ++j) acc[4 * g + j] = wmma_bf16(a1, bb[j], acc[4 * g + j]);
        }
        ASYNC_WAIT();
        __syncthreads();
    }
    const int l15 = lane & 15, hi = lane >> 4;
#pragma unroll
    for (int j = 0; j < 12; ++j) {
        int nrow = 16 * j + l15;     // 0..191
        int s    = nrow & 63;
        int grp  = nrow >> 6;        // 0=dt, 1=b, 2=c
#pragma unroll
        for (int v = 0; v < 8; ++v) {
            int m = m0 + v + hi * 8;
            size_t idx = (size_t)m * S_ + s;
            float a_v = acc[j][v];
            if (grp == 0) {
                float dtv = a_v + dt_b[s];
                dtv = (dtv > 20.f) ? dtv : log1pf(expf(dtv));   // softplus
                dtv += 1e-4f;
                decay[idx] = expf(-expf(a_log[s]) * dtv);
            } else if (grp == 1) {
                bval[idx] = tanhf(a_v + b_b[s]);
            } else {
                cval[idx] = tanhf(a_v + c_b[s]);
            }
        }
    }
}

// ---------------- chunked linear scan ----------------
__global__ void scan1_kernel(const float* __restrict__ decay, const float* __restrict__ bval,
                             float* __restrict__ P, float* __restrict__ E) {
    int idx = blockIdx.x * blockDim.x + threadIdx.x;   // (b*NCH + ch)*S + s
    if (idx >= B_ * NCH * S_) return;
    int s  = idx % S_;
    int ch = (idx / S_) % NCH;
    int b  = idx / (S_ * NCH);
    float p = 1.f, e = 0.f;
    size_t base = ((size_t)b * T_ + (size_t)ch * CHL) * S_ + s;
    for (int i = 0; i < CHL; ++i) {
        float d  = decay[base + (size_t)i * S_];
        float bb = bval[base + (size_t)i * S_];
        e = e * d + bb;
        p *= d;
    }
    P[idx] = p; E[idx] = e;
}

__global__ void scan2_kernel(const float* __restrict__ P, const float* __restrict__ E,
                             float* __restrict__ carry) {
    int idx = blockIdx.x * blockDim.x + threadIdx.x;   // b*S + s
    if (idx >= B_ * S_) return;
    int s = idx % S_;
    int b = idx / S_;
    float c = 0.f;
    for (int ch = 0; ch < NCH; ++ch) {
        size_t q = ((size_t)b * NCH + ch) * S_ + s;
        carry[q] = c;
        c = c * P[q] + E[q];
    }
}

__global__ void scan3_kernel(const float* __restrict__ decay, const float* __restrict__ bval,
                             const float* __restrict__ cval, const float* __restrict__ carry,
                             bf16_t* __restrict__ mix) {
    int idx = blockIdx.x * blockDim.x + threadIdx.x;
    if (idx >= B_ * NCH * S_) return;
    int s  = idx % S_;
    int ch = (idx / S_) % NCH;
    int b  = idx / (S_ * NCH);
    float st = carry[idx];
    size_t bt0 = (size_t)b * T_ + (size_t)ch * CHL;
    for (int i = 0; i < CHL; ++i) {
        size_t g = (bt0 + i) * S_ + s;
        st = st * decay[g] + bval[g];
        mix[(bt0 + i) * DS_ + D_ + s] = (bf16_t)(cval[g] * st);
    }
}

// ---------------- out GEMM: out = residual + mix @ out_w^T + out_b ----------------
__global__ void gemm_out_kernel(const bf16_t* __restrict__ A, const bf16_t* __restrict__ W,
                                const float* __restrict__ bias, const float* __restrict__ resid,
                                float* __restrict__ out) {
    __shared__ __align__(32) bf16_t sb[2 * 64 * 64];
    const int tid = threadIdx.x, lane = tid & 31, wave = tid >> 5;
    const int m0 = blockIdx.x * 128 + wave * 16;
    const int n0 = blockIdx.y * 64;
    v8f acc[4] = {v8f{0}, v8f{0}, v8f{0}, v8f{0}};
    gemm_mainloop(A, DS_, W, DS_, DS_, m0, n0, lane, tid, sb, acc);
    const int l15 = lane & 15, hi = lane >> 4;
#pragma unroll
    for (int j = 0; j < 4; ++j) {
        int n = n0 + 16 * j + l15;
#pragma unroll
        for (int v = 0; v < 8; ++v) {
            int m = m0 + v + hi * 8;
            size_t o = (size_t)m * D_ + n;
            out[o] = resid[o] + acc[j][v] + bias[n];
        }
    }
}

// ---------------- host orchestration ----------------
extern "C" void kernel_launch(void* const* d_in, const int* in_sizes, int n_in,
                              void* d_out, int out_size, void* d_ws, size_t ws_size,
                              hipStream_t stream) {
    const float* x      = (const float*)d_in[0];
    const float* norm_w = (const float*)d_in[1];
    const float* norm_b = (const float*)d_in[2];
    const float* in_w   = (const float*)d_in[3];
    const float* in_b   = (const float*)d_in[4];
    const float* conv_w = (const float*)d_in[5];
    const float* conv_b = (const float*)d_in[6];
    const float* dt_w   = (const float*)d_in[7];
    const float* dt_b   = (const float*)d_in[8];
    const float* b_w    = (const float*)d_in[9];
    const float* b_b    = (const float*)d_in[10];
    const float* c_w    = (const float*)d_in[11];
    const float* c_b    = (const float*)d_in[12];
    const float* a_log  = (const float*)d_in[13];
    const float* dvec   = (const float*)d_in[14];
    const float* out_w  = (const float*)d_in[15];
    const float* out_b  = (const float*)d_in[16];
    float* out = (float*)d_out;

    char* ws = (char*)d_ws;
    size_t off = 0;
    auto take = [&](size_t bytes) -> char* {
        char* p = ws + off;
        off += (bytes + 255) & ~(size_t)255;
        return p;
    };
    bf16_t* xn_bf   = (bf16_t*)take((size_t)M_TOT * D_ * 2);
    bf16_t* inw_bf  = (bf16_t*)take((size_t)TWO_D * D_ * 2);
    bf16_t* gw_bf   = (bf16_t*)take((size_t)192 * D_ * 2);     // packed [dt_w; b_w; c_w]
    bf16_t* outw_bf = (bf16_t*)take((size_t)D_ * DS_ * 2);
    bf16_t* xproj   = (bf16_t*)take((size_t)M_TOT * D_ * 2);
    bf16_t* z_bf    = (bf16_t*)take((size_t)M_TOT * D_ * 2);
    bf16_t* mix     = (bf16_t*)take((size_t)M_TOT * DS_ * 2);
    float*  decay   = (float*) take((size_t)M_TOT * S_ * 4);
    float*  bval    = (float*) take((size_t)M_TOT * S_ * 4);
    float*  cval    = (float*) take((size_t)M_TOT * S_ * 4);
    float*  Pc      = (float*) take((size_t)B_ * NCH * S_ * 4);
    float*  Ec      = (float*) take((size_t)B_ * NCH * S_ * 4);
    float*  carry   = (float*) take((size_t)B_ * NCH * S_ * 4);
    (void)ws_size; (void)n_in; (void)in_sizes; (void)out_size;

    // 1) weight repacks fp32 -> bf16
    {
        int n = TWO_D * D_;
        f2bf_kernel<<<(n + 255) / 256, 256, 0, stream>>>(in_w, inw_bf, n);
        n = S_ * D_;
        f2bf_kernel<<<(n + 255) / 256, 256, 0, stream>>>(dt_w, gw_bf, n);
        f2bf_kernel<<<(n + 255) / 256, 256, 0, stream>>>(b_w, gw_bf + (size_t)S_ * D_, n);
        f2bf_kernel<<<(n + 255) / 256, 256, 0, stream>>>(c_w, gw_bf + (size_t)2 * S_ * D_, n);
        n = D_ * DS_;
        f2bf_kernel<<<(n + 255) / 256, 256, 0, stream>>>(out_w, outw_bf, n);
    }

    // 2) LayerNorm
    ln_kernel<<<M_TOT, 256, 0, stream>>>(x, norm_w, norm_b, xn_bf);

    // 3) in_proj GEMM -> xproj, z
    gemm_in_kernel<<<dim3(M_TOT / 128, TWO_D / 64), 256, 0, stream>>>(
        xn_bf, inw_bf, in_b, xproj, z_bf);

    // 4) depthwise conv + tanh + *d -> mix[:, 0:D]
    {
        size_t n = (size_t)M_TOT * (D_ / 8);
        conv_kernel<<<(unsigned)((n + 255) / 256), 256, 0, stream>>>(
            xproj, conv_w, conv_b, dvec, mix);
    }

    // 5) gate GEMM + nonlinearity -> decay, bval, cval
    gates_kernel<<<M_TOT / 128, 256, 0, stream>>>(
        z_bf, gw_bf, dt_b, b_b, c_b, a_log, decay, bval, cval);

    // 6-8) chunked linear scan -> mix[:, D:D+S]
    {
        int n1 = B_ * NCH * S_;
        scan1_kernel<<<(n1 + 255) / 256, 256, 0, stream>>>(decay, bval, Pc, Ec);
        int n2 = B_ * S_;
        scan2_kernel<<<(n2 + 255) / 256, 256, 0, stream>>>(Pc, Ec, carry);
        scan3_kernel<<<(n1 + 255) / 256, 256, 0, stream>>>(decay, bval, cval, carry, mix);
    }

    // 9) out GEMM + bias + residual -> out (fp32)
    gemm_out_kernel<<<dim3(M_TOT / 128, D_ / 64), 256, 0, stream>>>(
        mix, outw_bf, out_b, x, out);
}
